// TransformerBlock_62251255988352
// MI455X (gfx1250) — compile-verified, matrix operand order checked by reading, not run on hardware
//
#include <hip/hip_runtime.h>

#define EMBED    1024
#define HEADS    16
#define HEAD_DIM 64
#define NEXP     8
#define HIDDEN   2752
#define SEQ      2048
#define BATCH    2
#define TOKENS   (BATCH*SEQ)   // 4096
#define NSLOTS   (TOKENS*2)    // 8192

#define BM 128
#define BN 64
#define BK 32
#define APAD 40     // LDS stride in halves (80B rows: 16B-aligned, conflict-free)
#define KSTR 72     // flash LDS stride in halves (144B)

typedef __attribute__((ext_vector_type(16))) __bf16 v16bf;
typedef __attribute__((ext_vector_type(8)))  float  v8f;
typedef __attribute__((ext_vector_type(4)))  unsigned int u32x4;
typedef __attribute__((ext_vector_type(8)))  int i32x8;
typedef __attribute__((ext_vector_type(4)))  int i32x4;

__device__ __forceinline__ unsigned short f2bf(float f) {
  unsigned int u = __float_as_uint(f);
  unsigned int r = (u + 0x7FFFu + ((u >> 16) & 1u)) >> 16;
  return (unsigned short)r;
}

__device__ __forceinline__ v8f wmma_bf16(uint4 a0, uint4 a1, uint4 b0, uint4 b1, v8f c) {
  union { uint4 q[2]; v16bf v; } A, B;
  A.q[0] = a0; A.q[1] = a1; B.q[0] = b0; B.q[1] = b1;
  return __builtin_amdgcn_wmma_f32_16x16x32_bf16(false, A.v, false, B.v, (short)0, c,
                                                 false, false);
}

__device__ __forceinline__ unsigned lds_off(const void* p) {
  return (unsigned)(unsigned long long)p;   // low 32 bits of LDS generic addr = DS offset
}

// Issue one 2-D TDM tile load: tile_d1 rows of tile_d0 bf16 elements, row stride
// `stride` elements in memory; LDS rows padded per (ic, ac) codes.
// ic: pad_interval code (3 => every 16 dwords = 64B row), ac: pad_amount code (3 => 4 dwords = 16B).
__device__ __forceinline__ void tdm_load_2d(unsigned ldsaddr, const void* gp,
                                            unsigned tile_d0, unsigned tile_d1,
                                            unsigned stride, unsigned ic, unsigned ac) {
  unsigned long long ga = (unsigned long long)gp;
  u32x4 g0;
  g0[0] = 1u;                                               // count=1, user mode
  g0[1] = ldsaddr;                                          // lds_addr
  g0[2] = (unsigned)ga;                                     // global_addr[31:0]
  g0[3] = ((unsigned)(ga >> 32) & 0x01FFFFFFu) | 0x80000000u; // addr[56:32] | type=2
  i32x8 g1;
  g1[0] = (int)((1u << 16) | (1u << 20) | (ic << 22) | (ac << 25)); // 2B data, pad_enable
  g1[1] = (int)((stride & 0xFFFFu) << 16);                  // tensor_dim0 = stride (lo16)
  g1[2] = (int)((stride >> 16) | ((tile_d1 & 0xFFFFu) << 16)); // dim0 hi | tensor_dim1 lo
  g1[3] = (int)(tile_d0 << 16);                             // tensor_dim1 hi=0 | tile_dim0
  g1[4] = (int)tile_d1;                                     // tile_dim1 (tile_dim2=0)
  g1[5] = (int)stride;                                      // tensor_dim0_stride lo32
  g1[6] = 0;
  g1[7] = 0;
  i32x4 z4 = {0, 0, 0, 0};
  i32x8 z8 = {0, 0, 0, 0, 0, 0, 0, 0};
  __builtin_amdgcn_tensor_load_to_lds(g0, g1, z4, z4, z8, 0);
}

// ---------------------------------------------------------------- utilities
// fp32 [R,C] -> bf16 transposed [C,R]; grid (C/32, R/32, Z), block (32,8)
__global__ __launch_bounds__(256) void k_transpose_bf16(const float* __restrict__ in,
                                                        unsigned short* __restrict__ outp,
                                                        int R, int C) {
  __shared__ float tile[32][33];
  const float* src = in + (size_t)blockIdx.z * R * C;
  unsigned short* dst = outp + (size_t)blockIdx.z * R * C;
  int c0 = blockIdx.x * 32, r0 = blockIdx.y * 32;
  int x = threadIdx.x, y = threadIdx.y;
  #pragma unroll
  for (int i = 0; i < 4; i++)
    tile[y + 8 * i][x] = src[(size_t)(r0 + y + 8 * i) * C + c0 + x];
  __syncthreads();
  #pragma unroll
  for (int i = 0; i < 4; i++)
    dst[(size_t)(c0 + y + 8 * i) * R + r0 + x] = f2bf(tile[x][y + 8 * i]);
}

__global__ __launch_bounds__(256) void k_rmsnorm_bf16(const float* __restrict__ X,
                                                      const float* __restrict__ gamma,
                                                      unsigned short* __restrict__ O) {
  __shared__ float red[256];
  int row = blockIdx.x, tid = threadIdx.x;
  const float* xr = X + (size_t)row * EMBED;
  float4 v = *(const float4*)(xr + tid * 4);
  red[tid] = v.x * v.x + v.y * v.y + v.z * v.z + v.w * v.w;
  __syncthreads();
  for (int s = 128; s > 0; s >>= 1) { if (tid < s) red[tid] += red[tid + s]; __syncthreads(); }
  float r = rsqrtf(red[0] * (1.0f / EMBED) + 1e-6f);
  float4 g = *(const float4*)(gamma + tid * 4);
  unsigned short* op = O + (size_t)row * EMBED + tid * 4;
  op[0] = f2bf(v.x * r * g.x); op[1] = f2bf(v.y * r * g.y);
  op[2] = f2bf(v.z * r * g.z); op[3] = f2bf(v.w * r * g.w);
}

__global__ void k_add(const float* __restrict__ a, const float* __restrict__ b,
                      float* __restrict__ c, long n) {
  long i = (long)blockIdx.x * blockDim.x + threadIdx.x;
  long st = (long)gridDim.x * blockDim.x;
  for (; i < n; i += st) c[i] = a[i] + b[i];
}

// ---------------------------------------------------------------- generic WMMA GEMM
// C[M,N] = A[M,K] @ Bt[N,K]^T. A row-major bf16, Bt transposed bf16. TDM-staged,
// double-buffered, software-pipelined. M%128==0, N%64==0, K%32==0.
__global__ __launch_bounds__(256) void k_gemm_bf16(const unsigned short* __restrict__ A,
                                                   const unsigned short* __restrict__ Bt,
                                                   float* __restrict__ C, int K, int N) {
  __shared__ alignas(16) unsigned short As[2][BM * APAD];
  __shared__ alignas(16) unsigned short Bs[2][BN * APAD];
  int tid = threadIdx.x, lane = tid & 31, wv = tid >> 5;
  int hi8 = (lane >> 4) << 3, m16 = lane & 15;
  int row0 = blockIdx.y * BM, col0 = blockIdx.x * BN;
  const unsigned short* Abase = A + (size_t)row0 * K;
  const unsigned short* Bbase = Bt + (size_t)col0 * K;
  v8f acc[4] = {};
  int nk = K / BK;
  if (tid < 32)      tdm_load_2d(lds_off(Bs[0]), Bbase, BK, BN, K, 3, 3);
  else if (tid < 64) tdm_load_2d(lds_off(As[0]), Abase, BK, BM, K, 3, 3);
  for (int i = 0; i < nk; i++) {
    int cur = i & 1, nb = (i + 1) & 1;
    bool nxt = (i + 1 < nk);
    if (tid < 32) {
      if (nxt) { tdm_load_2d(lds_off(Bs[nb]), Bbase + (i + 1) * BK, BK, BN, K, 3, 3);
                 __builtin_amdgcn_s_wait_tensorcnt(1); }
      else     __builtin_amdgcn_s_wait_tensorcnt(0);
    } else if (tid < 64) {
      if (nxt) { tdm_load_2d(lds_off(As[nb]), Abase + (i + 1) * BK, BK, BM, K, 3, 3);
                 __builtin_amdgcn_s_wait_tensorcnt(1); }
      else     __builtin_amdgcn_s_wait_tensorcnt(0);
    }
    __syncthreads();
    const unsigned short* as = As[cur];
    const unsigned short* bs = Bs[cur];
    uint4 a0 = *(const uint4*)(as + (wv * 16 + m16) * APAD + hi8);
    uint4 a1 = *(const uint4*)(as + (wv * 16 + m16) * APAD + 16 + hi8);
    uint4 bf[4][2];
    #pragma unroll
    for (int nt = 0; nt < 4; nt++) {
      const unsigned short* bp = bs + (nt * 16 + m16) * APAD;
      bf[nt][0] = *(const uint4*)(bp + hi8);
      bf[nt][1] = *(const uint4*)(bp + 16 + hi8);
    }
    #pragma unroll
    for (int nt = 0; nt < 4; nt++)
      acc[nt] = wmma_bf16(a0, a1, bf[nt][0], bf[nt][1], acc[nt]);
    __syncthreads();
  }
  int mrow = row0 + wv * 16 + hi8;
  #pragma unroll
  for (int nt = 0; nt < 4; nt++)
    #pragma unroll
    for (int j = 0; j < 8; j++)
      C[(size_t)(mrow + j) * N + col0 + nt * 16 + m16] = acc[nt][j];
}

// ---------------------------------------------------------------- RoPE + [B,H,S,D] pack
__global__ void k_rope_pack(const float* __restrict__ Xf, unsigned short* __restrict__ O, int doRope) {
  int idx = blockIdx.x * blockDim.x + threadIdx.x;
  int t = idx >> 10, c = idx & 1023;
  int b = t >> 11, s = t & (SEQ - 1);
  int hd = c >> 6, d = c & 63;
  float v = Xf[idx], res = v;
  if (doRope) {
    int j = d & 31;
    float invf = __expf(-(float)j * 0.28782314f);   // ln(10000)/32
    float ang = (float)s * invf;
    float cs = __cosf(ang), sn = __sinf(ang);
    float other = (d < 32) ? -Xf[idx + 32] : Xf[idx - 32];
    res = v * cs + other * sn;
  }
  O[(((size_t)(b * HEADS + hd)) * SEQ + s) * HEAD_DIM + d] = f2bf(res);
}

// ---------------------------------------------------------------- flash attention (WMMA)
__global__ __launch_bounds__(256) void k_flash(const unsigned short* __restrict__ Qb,
                                               const unsigned short* __restrict__ Kb,
                                               const unsigned short* __restrict__ Vb,
                                               unsigned short* __restrict__ Ob) {
  __shared__ alignas(16) unsigned short Ks[64 * KSTR];
  __shared__ alignas(16) unsigned short Vt[64 * KSTR];
  __shared__ alignas(16) unsigned short Pt[8 * 16 * KSTR];
  int tid = threadIdx.x, lane = tid & 31, wv = tid >> 5;
  int hi8 = (lane >> 4) << 3, m16 = lane & 15;
  int qblk = blockIdx.x, bh = blockIdx.y;
  const size_t hb = (size_t)bh * SEQ * HEAD_DIM;
  int qbase = qblk * 128 + wv * 16;
  uint4 qa[2][2];
  #pragma unroll
  for (int s = 0; s < 2; s++) {
    const unsigned short* qp = Qb + hb + (size_t)(qbase + m16) * HEAD_DIM + s * 32 + hi8;
    qa[s][0] = *(const uint4*)qp;
    qa[s][1] = *(const uint4*)(qp + 16);
  }
  v8f o[4] = {};
  float mrun[8], lrun[8];
  #pragma unroll
  for (int j = 0; j < 8; j++) { mrun[j] = -1e30f; lrun[j] = 0.f; }
  int kbmax = qblk * 2 + 1;
  for (int kb = 0; kb <= kbmax; kb++) {
    __syncthreads();
    if (tid < 32) {   // K tile 64x64, row=128B (interval code 4), pad 16B -> KSTR
      tdm_load_2d(lds_off(Ks), Kb + hb + (size_t)kb * 64 * HEAD_DIM, 64, 64, 64, 4, 3);
      __builtin_amdgcn_s_wait_tensorcnt(0);
    }
    { int r = tid >> 2, dc = (tid & 3) << 4;   // V transposed staging
      const unsigned short* g = Vb + hb + (size_t)(kb * 64 + r) * HEAD_DIM + dc;
      #pragma unroll
      for (int i = 0; i < 16; i++) Vt[(dc + i) * KSTR + r] = g[i]; }
    __syncthreads();
    v8f sc[4];
    #pragma unroll
    for (int nt = 0; nt < 4; nt++) {
      uint4 b0[2], b1[2];
      #pragma unroll
      for (int s = 0; s < 2; s++) {
        const unsigned short* bp = Ks + (nt * 16 + m16) * KSTR + s * 32 + hi8;
        b0[s] = *(const uint4*)bp; b1[s] = *(const uint4*)(bp + 16);
      }
      v8f c = {};
      c = wmma_bf16(qa[0][0], qa[0][1], b0[0], b1[0], c);
      c = wmma_bf16(qa[1][0], qa[1][1], b0[1], b1[1], c);
      sc[nt] = c;
    }
    #pragma unroll
    for (int j = 0; j < 8; j++) {
      int qrow = qbase + hi8 + j;
      float tmax = -1e30f;
      #pragma unroll
      for (int nt = 0; nt < 4; nt++) {
        int key = kb * 64 + nt * 16 + m16;
        float v = sc[nt][j] * 0.125f;
        if (key > qrow) v = -1e30f;
        sc[nt][j] = v;
        tmax = fmaxf(tmax, v);
      }
      #pragma unroll
      for (int d = 1; d < 16; d <<= 1) tmax = fmaxf(tmax, __shfl_xor(tmax, d, 32));
      float mnew = fmaxf(mrun[j], tmax);
      float corr = __expf(mrun[j] - mnew);
      float psum = 0.f;
      #pragma unroll
      for (int nt = 0; nt < 4; nt++) {
        float s0 = sc[nt][j];
        float p = (s0 <= -1e29f) ? 0.f : __expf(s0 - mnew);
        sc[nt][j] = p; psum += p;
      }
      #pragma unroll
      for (int d = 1; d < 16; d <<= 1) psum += __shfl_xor(psum, d, 32);
      lrun[j] = lrun[j] * corr + psum;
      mrun[j] = mnew;
      #pragma unroll
      for (int nt = 0; nt < 4; nt++) {
        o[nt][j] *= corr;
        Pt[(wv * 16 + hi8 + j) * KSTR + nt * 16 + m16] = f2bf(sc[nt][j]);
      }
    }
    __syncthreads();
    #pragma unroll
    for (int s = 0; s < 2; s++) {
      const unsigned short* ap = Pt + (wv * 16 + m16) * KSTR + s * 32 + hi8;
      uint4 a0 = *(const uint4*)ap, a1 = *(const uint4*)(ap + 16);
      #pragma unroll
      for (int nt = 0; nt < 4; nt++) {
        const unsigned short* bp = Vt + (nt * 16 + m16) * KSTR + s * 32 + hi8;
        uint4 b0 = *(const uint4*)bp, b1 = *(const uint4*)(bp + 16);
        o[nt] = wmma_bf16(a0, a1, b0, b1, o[nt]);
      }
    }
  }
  int b = bh >> 4, hd = bh & 15;
  #pragma unroll
  for (int j = 0; j < 8; j++) {
    int qrow = qbase + hi8 + j;
    float inv = 1.f / lrun[j];
    #pragma unroll
    for (int nt = 0; nt < 4; nt++)
      Ob[((size_t)(b * SEQ + qrow)) * EMBED + hd * HEAD_DIM + nt * 16 + m16] =
          f2bf(o[nt][j] * inv);
  }
}

// ---------------------------------------------------------------- router + grouping
__global__ __launch_bounds__(256) void k_router(const float* __restrict__ H,
                                                const float* __restrict__ g2,
                                                const float* __restrict__ Wr,
                                                int* __restrict__ topi, float* __restrict__ topw) {
  __shared__ float red[256];
  __shared__ float acc8[256 * 8];
  int t = blockIdx.x, tid = threadIdx.x;
  const float* h = H + (size_t)t * EMBED;
  float4 v = *(const float4*)(h + tid * 4);
  red[tid] = v.x * v.x + v.y * v.y + v.z * v.z + v.w * v.w;
  __syncthreads();
  for (int s = 128; s > 0; s >>= 1) { if (tid < s) red[tid] += red[tid + s]; __syncthreads(); }
  float r = rsqrtf(red[0] * (1.0f / EMBED) + 1e-6f);
  float a[8] = {0, 0, 0, 0, 0, 0, 0, 0};
  float xl[4] = {v.x, v.y, v.z, v.w};
  #pragma unroll
  for (int i = 0; i < 4; i++) {
    int d = tid * 4 + i;
    float xn = xl[i] * r * g2[d];
    #pragma unroll
    for (int x = 0; x < 8; x++) a[x] += xn * Wr[d * 8 + x];
  }
  #pragma unroll
  for (int x = 0; x < 8; x++) acc8[tid * 8 + x] = a[x];
  __syncthreads();
  if (tid < 8) {
    float s = 0.f;
    for (int i = 0; i < 256; i++) s += acc8[i * 8 + tid];
    red[tid] = s;
  }
  __syncthreads();
  if (tid == 0) {
    float l[8];
    for (int x = 0; x < 8; x++) l[x] = red[x];
    int i0 = 0;
    for (int x = 1; x < 8; x++) if (l[x] > l[i0]) i0 = x;
    int i1 = (i0 == 0) ? 1 : 0;
    for (int x = 0; x < 8; x++) if (x != i0 && l[x] > l[i1]) i1 = x;
    float e1 = __expf(l[i1] - l[i0]);
    float inv = 1.f / (1.f + e1);
    topi[2 * t] = i0; topi[2 * t + 1] = i1;
    topw[2 * t] = inv; topw[2 * t + 1] = e1 * inv;
  }
}

__global__ void k_count(const int* __restrict__ topi, int* __restrict__ cnt) {
  int i = blockIdx.x * blockDim.x + threadIdx.x;
  if (i < NSLOTS) atomicAdd(&cnt[topi[i]], 1);
}

__global__ void k_offsets(const int* __restrict__ cnt, int* __restrict__ offs) {
  if (threadIdx.x == 0) {
    int a = 0;
    for (int x = 0; x < NEXP; x++) { offs[x] = a; a += cnt[x]; }
    offs[NEXP] = a;
  }
}

__global__ __launch_bounds__(256) void k_assign(const int* __restrict__ topi,
                                                const int* __restrict__ offs,
                                                int* __restrict__ map, int* __restrict__ entry_of) {
  __shared__ int sc[256];
  int e = blockIdx.x, tid = threadIdx.x;
  int begin = tid * (NSLOTS / 256);
  int c = 0;
  for (int i = 0; i < NSLOTS / 256; i++) c += (topi[begin + i] == e);
  sc[tid] = c;
  __syncthreads();
  for (int s = 1; s < 256; s <<= 1) {
    int v = (tid >= s) ? sc[tid - s] : 0;
    __syncthreads();
    sc[tid] += v;
    __syncthreads();
  }
  int pos = offs[e] + sc[tid] - c;
  for (int i = 0; i < NSLOTS / 256; i++) {
    int slot = begin + i;
    if (topi[slot] == e) { map[pos] = slot >> 1; entry_of[slot] = pos; pos++; }
  }
}

// ---------------------------------------------------------------- MoE expert GEMMs
__global__ __launch_bounds__(256) void k_moe_gu(const unsigned short* __restrict__ Xbf,
                                                const unsigned short* __restrict__ Wgt,
                                                const unsigned short* __restrict__ Wut,
                                                unsigned short* __restrict__ GU,
                                                const int* __restrict__ map,
                                                const int* __restrict__ offs) {
  __shared__ alignas(16) unsigned short As[BM * APAD];
  __shared__ alignas(16) unsigned short Bg[2][BN * APAD];
  __shared__ alignas(16) unsigned short Bu[2][BN * APAD];
  int e = blockIdx.z;
  int base = offs[e], cnt = offs[e + 1] - base;
  int rb = blockIdx.y;
  if (rb * BM >= cnt) return;
  int tid = threadIdx.x, lane = tid & 31, wv = tid >> 5;
  int hi8 = (lane >> 4) << 3, m16 = lane & 15;
  int col0 = blockIdx.x * BN;
  const unsigned short* Gbase = Wgt + (size_t)e * HIDDEN * EMBED + (size_t)col0 * EMBED;
  const unsigned short* Ubase = Wut + (size_t)e * HIDDEN * EMBED + (size_t)col0 * EMBED;
  v8f accg[4] = {}, accu[4] = {};
  int nk = EMBED / BK;
  if (tid < 32) {
    tdm_load_2d(lds_off(Bg[0]), Gbase, BK, BN, EMBED, 3, 3);
    tdm_load_2d(lds_off(Bu[0]), Ubase, BK, BN, EMBED, 3, 3);
  }
  for (int i = 0; i < nk; i++) {
    int cur = i & 1, nb = (i + 1) & 1;
    bool nxt = (i + 1 < nk);
    int k0 = i * BK;
    { int r = tid >> 1, co = (tid & 1) << 4;    // gathered A staging
      int lr = rb * BM + r;
      uint4 v0 = {0, 0, 0, 0}, v1 = {0, 0, 0, 0};
      if (lr < cnt) {
        int tok = map[base + lr];
        const uint4* g = (const uint4*)(Xbf + (size_t)tok * EMBED + k0 + co);
        v0 = g[0]; v1 = g[1];
      }
      uint4* s = (uint4*)(As + r * APAD + co); s[0] = v0; s[1] = v1; }
    if (tid < 32) {
      if (nxt) {
        tdm_load_2d(lds_off(Bg[nb]), Gbase + (i + 1) * BK, BK, BN, EMBED, 3, 3);
        tdm_load_2d(lds_off(Bu[nb]), Ubase + (i + 1) * BK, BK, BN, EMBED, 3, 3);
        __builtin_amdgcn_s_wait_tensorcnt(2);
      } else __builtin_amdgcn_s_wait_tensorcnt(0);
    }
    __syncthreads();
    uint4 a0 = *(const uint4*)(As + (wv * 16 + m16) * APAD + hi8);
    uint4 a1 = *(const uint4*)(As + (wv * 16 + m16) * APAD + 16 + hi8);
    uint4 fg[4][2], fu[4][2];
    #pragma unroll
    for (int nt = 0; nt < 4; nt++) {
      const unsigned short* bp = Bg[cur] + (nt * 16 + m16) * APAD;
      fg[nt][0] = *(const uint4*)(bp + hi8);
      fg[nt][1] = *(const uint4*)(bp + 16 + hi8);
      const unsigned short* up = Bu[cur] + (nt * 16 + m16) * APAD;
      fu[nt][0] = *(const uint4*)(up + hi8);
      fu[nt][1] = *(const uint4*)(up + 16 + hi8);
    }
    #pragma unroll
    for (int nt = 0; nt < 4; nt++) {
      accg[nt] = wmma_bf16(a0, a1, fg[nt][0], fg[nt][1], accg[nt]);
      accu[nt] = wmma_bf16(a0, a1, fu[nt][0], fu[nt][1], accu[nt]);
    }
    __syncthreads();
  }
  #pragma unroll
  for (int nt = 0; nt < 4; nt++)
    #pragma unroll
    for (int j = 0; j < 8; j++) {
      int lr = rb * BM + wv * 16 + hi8 + j;
      if (lr < cnt) {
        float g = accg[nt][j], u = accu[nt][j];
        float sg = g / (1.f + __expf(-g));
        GU[(size_t)(base + lr) * HIDDEN + col0 + nt * 16 + m16] = f2bf(sg * u);
      }
    }
}

__global__ __launch_bounds__(256) void k_moe_down(const unsigned short* __restrict__ GU,
                                                  const unsigned short* __restrict__ Wdt,
                                                  float* __restrict__ EO,
                                                  const int* __restrict__ offs) {
  __shared__ alignas(16) unsigned short As[BM * APAD];
  __shared__ alignas(16) unsigned short Bs[2][BN * APAD];
  int e = blockIdx.z;
  int base = offs[e], cnt = offs[e + 1] - base;
  int rb = blockIdx.y;
  if (rb * BM >= cnt) return;
  int tid = threadIdx.x, lane = tid & 31, wv = tid >> 5;
  int hi8 = (lane >> 4) << 3, m16 = lane & 15;
  int col0 = blockIdx.x * BN;
  const unsigned short* Bbase = Wdt + (size_t)e * EMBED * HIDDEN + (size_t)col0 * HIDDEN;
  v8f acc[4] = {};
  int nk = HIDDEN / BK;
  if (tid < 32) tdm_load_2d(lds_off(Bs[0]), Bbase, BK, BN, HIDDEN, 3, 3);
  for (int i = 0; i < nk; i++) {
    int cur = i & 1, nb = (i + 1) & 1;
    bool nxt = (i + 1 < nk);
    int k0 = i * BK;
    { int r = tid >> 1, co = (tid & 1) << 4;
      int lr = rb * BM + r;
      uint4 v0 = {0, 0, 0, 0}, v1 = {0, 0, 0, 0};
      if (lr < cnt) {
        const uint4* g = (const uint4*)(GU + (size_t)(base + lr) * HIDDEN + k0 + co);
        v0 = g[0]; v1 = g[1];
      }
      uint4* s = (uint4*)(As + r * APAD + co); s[0] = v0; s[1] = v1; }
    if (tid < 32) {
      if (nxt) { tdm_load_2d(lds_off(Bs[nb]), Bbase + (i + 1) * BK, BK, BN, HIDDEN, 3, 3);
                 __builtin_amdgcn_s_wait_tensorcnt(1); }
      else __builtin_amdgcn_s_wait_tensorcnt(0);
    }
    __syncthreads();
    uint4 a0 = *(const uint4*)(As + (wv * 16 + m16) * APAD + hi8);
    uint4 a1 = *(const uint4*)(As + (wv * 16 + m16) * APAD + 16 + hi8);
    uint4 bf[4][2];
    #pragma unroll
    for (int nt = 0; nt < 4; nt++) {
      const unsigned short* bp = Bs[cur] + (nt * 16 + m16) * APAD;
      bf[nt][0] = *(const uint4*)(bp + hi8);
      bf[nt][1] = *(const uint4*)(bp + 16 + hi8);
    }
    #pragma unroll
    for (int nt = 0; nt < 4; nt++)
      acc[nt] = wmma_bf16(a0, a1, bf[nt][0], bf[nt][1], acc[nt]);
    __syncthreads();
  }
  #pragma unroll
  for (int nt = 0; nt < 4; nt++)
    #pragma unroll
    for (int j = 0; j < 8; j++) {
      int lr = rb * BM + wv * 16 + hi8 + j;
      if (lr < cnt)
        EO[(size_t)(base + lr) * EMBED + col0 + nt * 16 + m16] = acc[nt][j];
    }
}

__global__ void k_combine(const float* __restrict__ H, const float* __restrict__ EO,
                          const int* __restrict__ entry_of, const float* __restrict__ topw,
                          float* __restrict__ out, long n) {
  long i = (long)blockIdx.x * blockDim.x + threadIdx.x;
  long st = (long)gridDim.x * blockDim.x;
  for (; i < n; i += st) {
    long t = i >> 10, nn = i & 1023;
    int r0 = entry_of[2 * t], r1 = entry_of[2 * t + 1];
    out[i] = H[i] + topw[2 * t] * EO[(size_t)r0 * EMBED + nn]
                  + topw[2 * t + 1] * EO[(size_t)r1 * EMBED + nn];
  }
}

// ---------------------------------------------------------------- launcher
extern "C" void kernel_launch(void* const* d_in, const int* in_sizes, int n_in,
                              void* d_out, int out_size, void* d_ws, size_t ws_size,
                              hipStream_t stream) {
  (void)in_sizes; (void)n_in; (void)out_size; (void)ws_size;
  const float* x  = (const float*)d_in[0];
  const float* wq = (const float*)d_in[1];
  const float* wk = (const float*)d_in[2];
  const float* wv = (const float*)d_in[3];
  const float* wo = (const float*)d_in[4];
  const float* wr = (const float*)d_in[5];
  const float* wg = (const float*)d_in[6];
  const float* wu = (const float*)d_in[7];
  const float* wd = (const float*)d_in[8];
  const float* g1 = (const float*)d_in[9];
  const float* g2 = (const float*)d_in[10];
  float* out = (float*)d_out;

  char* ws = (char*)d_ws;
  size_t off = 0;
  auto alloc = [&](size_t b) -> char* {
    char* p = ws + off;
    off = (off + b + 255) & ~(size_t)255;
    return p;
  };
  const size_t SZ_E2  = (size_t)EMBED * EMBED * 2;
  const size_t SZ_MOE = (size_t)NEXP * EMBED * HIDDEN * 2;
  unsigned short* wq_bf = (unsigned short*)alloc(SZ_E2);    // transposed [N,K] bf16
  unsigned short* wk_bf = (unsigned short*)alloc(SZ_E2);
  unsigned short* wv_bf = (unsigned short*)alloc(SZ_E2);
  unsigned short* wo_bf = (unsigned short*)alloc(SZ_E2);
  unsigned short* wg_bf = (unsigned short*)alloc(SZ_MOE);   // [X][HIDDEN][EMBED]
  unsigned short* wu_bf = (unsigned short*)alloc(SZ_MOE);
  unsigned short* wd_bf = (unsigned short*)alloc(SZ_MOE);   // [X][EMBED][HIDDEN]
  char* regA = alloc(5ull * TOKENS * EMBED * 2);
  unsigned short* xn_bf = (unsigned short*)regA;
  unsigned short* q_bf  = (unsigned short*)(regA + 1ull * TOKENS * EMBED * 2);
  unsigned short* k_bf  = (unsigned short*)(regA + 2ull * TOKENS * EMBED * 2);
  unsigned short* v_bf  = (unsigned short*)(regA + 3ull * TOKENS * EMBED * 2);
  unsigned short* attnA = (unsigned short*)(regA + 4ull * TOKENS * EMBED * 2);
  float* eo = (float*)regA;                      // reuse: NSLOTS*EMBED*4 = 32MB <= 40MB
  char* regB = alloc(3ull * TOKENS * EMBED * 4);
  float* qf = (float*)regB;
  float* kf = (float*)(regB + 1ull * TOKENS * EMBED * 4);
  float* vf = (float*)(regB + 2ull * TOKENS * EMBED * 4);
  unsigned short* gu_bf = (unsigned short*)regB; // reuse: NSLOTS*HIDDEN*2 = 45MB <= 48MB
  float* proj  = (float*)alloc((size_t)TOKENS * EMBED * 4);
  float* hbuf  = (float*)alloc((size_t)TOKENS * EMBED * 4);
  unsigned short* hn_bf = (unsigned short*)alloc((size_t)TOKENS * EMBED * 2);
  int*   topi = (int*)alloc(NSLOTS * 4);
  float* topw = (float*)alloc(NSLOTS * 4);
  int*   cnt  = (int*)alloc(NEXP * 4);
  int*   offs = (int*)alloc((NEXP + 1) * 4);
  int*   map  = (int*)alloc(NSLOTS * 4);
  int*   eof  = (int*)alloc(NSLOTS * 4);

  // 1) weights -> bf16, transposed (B tiles become contiguous rows for TDM)
  dim3 tb(32, 8);
  k_transpose_bf16<<<dim3(32, 32, 1), tb, 0, stream>>>(wq, wq_bf, EMBED, EMBED);
  k_transpose_bf16<<<dim3(32, 32, 1), tb, 0, stream>>>(wk, wk_bf, EMBED, EMBED);
  k_transpose_bf16<<<dim3(32, 32, 1), tb, 0, stream>>>(wv, wv_bf, EMBED, EMBED);
  k_transpose_bf16<<<dim3(32, 32, 1), tb, 0, stream>>>(wo, wo_bf, EMBED, EMBED);
  k_transpose_bf16<<<dim3(HIDDEN / 32, EMBED / 32, NEXP), tb, 0, stream>>>(wg, wg_bf, EMBED, HIDDEN);
  k_transpose_bf16<<<dim3(HIDDEN / 32, EMBED / 32, NEXP), tb, 0, stream>>>(wu, wu_bf, EMBED, HIDDEN);
  k_transpose_bf16<<<dim3(EMBED / 32, HIDDEN / 32, NEXP), tb, 0, stream>>>(wd, wd_bf, HIDDEN, EMBED);

  // 2) pre-attn RMSNorm
  k_rmsnorm_bf16<<<TOKENS, 256, 0, stream>>>(x, g1, xn_bf);

  // 3) QKV projections (WMMA + TDM pipeline)
  dim3 gp(EMBED / BN, TOKENS / BM);
  k_gemm_bf16<<<gp, 256, 0, stream>>>(xn_bf, wq_bf, qf, EMBED, EMBED);
  k_gemm_bf16<<<gp, 256, 0, stream>>>(xn_bf, wk_bf, kf, EMBED, EMBED);
  k_gemm_bf16<<<gp, 256, 0, stream>>>(xn_bf, wv_bf, vf, EMBED, EMBED);

  // 4) RoPE + pack to [B,H,S,D] bf16
  int nel = TOKENS * EMBED;
  k_rope_pack<<<nel / 256, 256, 0, stream>>>(qf, q_bf, 1);
  k_rope_pack<<<nel / 256, 256, 0, stream>>>(kf, k_bf, 1);
  k_rope_pack<<<nel / 256, 256, 0, stream>>>(vf, v_bf, 0);

  // 5) flash attention
  k_flash<<<dim3(SEQ / 128, BATCH * HEADS), 256, 0, stream>>>(q_bf, k_bf, v_bf, attnA);

  // 6) output projection + residual
  k_gemm_bf16<<<gp, 256, 0, stream>>>(attnA, wo_bf, proj, EMBED, EMBED);
  k_add<<<4096, 256, 0, stream>>>(x, proj, hbuf, (long)TOKENS * EMBED);

  // 7) pre-MoE RMSNorm + router + deterministic grouping
  k_rmsnorm_bf16<<<TOKENS, 256, 0, stream>>>(hbuf, g2, hn_bf);
  k_router<<<TOKENS, 256, 0, stream>>>(hbuf, g2, wr, topi, topw);
  (void)hipMemsetAsync(cnt, 0, NEXP * 4, stream);
  k_count<<<NSLOTS / 256, 256, 0, stream>>>(topi, cnt);
  k_offsets<<<1, 1, 0, stream>>>(cnt, offs);
  k_assign<<<NEXP, 256, 0, stream>>>(topi, offs, map, eof);

  // 8) expert GEMMs (WMMA + TDM pipeline): fused gate/up + SiLU*U, then down proj
  k_moe_gu<<<dim3(HIDDEN / BN, NSLOTS / BM, NEXP), 256, 0, stream>>>(hn_bf, wg_bf, wu_bf,
                                                                    gu_bf, map, offs);
  k_moe_down<<<dim3(EMBED / BN, NSLOTS / BM, NEXP), 256, 0, stream>>>(gu_bf, wd_bf, eo, offs);

  // 9) weighted combine + residual
  k_combine<<<4096, 256, 0, stream>>>(hbuf, eo, eof, topw, out, (long)TOKENS * EMBED);
}
